// LinearPerformerAttentionMusic_29875792511831
// MI455X (gfx1250) — matrix operand order repaired and checked by reading, not compiled, not run on hardware
//
#include <hip/hip_runtime.h>
#include <hip/hip_bf16.h>
#include <math.h>

// ---------------------------------------------------------------------------
// MI455X (gfx1250) LinearPerformerAttentionMusic block.
// Every GEMM-shaped stage runs on v_wmma_f32_16x16x32_bf16 (wave32). All LDS
// tiles are stored K-contiguous so each WMMA fragment is exactly two
// ds_load_b128 ops (no scalar u16 assembly, no v_mov_b16 packing).
// Causal cumulative attention is chunked (T=64) so the scan becomes GEMMs.
// ---------------------------------------------------------------------------

typedef __bf16 bf16_t;
typedef __attribute__((ext_vector_type(16))) __bf16 v16bf;
typedef __attribute__((ext_vector_type(8)))  float  v8f;

#define DEV __device__ __forceinline__

DEV v8f wmma_bf16(v16bf a, v16bf b, v8f c) {
  // (neg_a, A, neg_b, B, c_mod, C, reuse_a, reuse_b)
  return __builtin_amdgcn_wmma_f32_16x16x32_bf16(false, a, false, b, (short)0, c,
                                                 false, false);
}

DEV v8f v8f_zero() {
  v8f z;
#pragma unroll
  for (int e = 0; e < 8; ++e) z[e] = 0.0f;
  return z;
}

// A fragment (16x32 bf16) from a K-contiguous [m][k] tile.
// Lane L: row m=L%16, kh=(L>=16?8:0). Needs halfs [kh..kh+7] and [kh+16..kh+23]
// -> two 16B LDS loads. ld must be a multiple of 8 halfs.
DEV v16bf frag_a_rm(const bf16_t* p, int ld) {
  const int lane = threadIdx.x & 31;
  const int m = lane & 15;
  const int kh = (lane & 16) ? 8 : 0;
  union { uint4 q[2]; v16bf v; } u;
  u.q[0] = *(const uint4*)&p[m * ld + kh];
  u.q[1] = *(const uint4*)&p[m * ld + kh + 16];
  return u.v;
}

// B fragment (32x16 bf16) from a K-contiguous [n][k] tile.
// Lane L: col n=L%16, kb=(L>=16?8:0). Needs halfs [kb..kb+7] and [kb+16..kb+23]
// -> two 16B LDS loads.
DEV v16bf frag_b_nk(const bf16_t* p, int ld) {
  const int lane = threadIdx.x & 31;
  const int n = lane & 15;
  const int kb = (lane & 16) ? 8 : 0;
  union { uint4 q[2]; v16bf v; } u;
  u.q[0] = *(const uint4*)&p[n * ld + kb];
  u.q[1] = *(const uint4*)&p[n * ld + kb + 16];
  return u.v;
}

// ---------------------------------------------------------------------------
// f32 -> bf16 conversion (+ proj transpose variant)
// ---------------------------------------------------------------------------
__global__ __launch_bounds__(256) void cvt_kernel(const float* __restrict__ in,
                                                  bf16_t* __restrict__ out, int n) {
  const int i = blockIdx.x * 256 + threadIdx.x;
  if (i < n) out[i] = (bf16_t)in[i];
}

// proj (h,hd,f) f32 -> projT (h,f,hd) bf16
__global__ __launch_bounds__(256) void cvt_tr_proj_kernel(
    const float* __restrict__ in, bf16_t* __restrict__ out) {
  const int i = blockIdx.x * 256 + threadIdx.x;  // 8*64*128 = 65536
  const int h = i >> 13, rem = i & 8191;
  const int hd = rem >> 7, f = rem & 127;
  out[((size_t)h * 128 + f) * 64 + hd] = (bf16_t)in[i];
}

// ---------------------------------------------------------------------------
// Generic C = A(MxK) * W(NxK)^T  (both bf16 row-major), 128x64 block tile,
// 8 waves each computing a 32x32 quadrant (2x2 WMMA frags), K step 32.
// EPI: 1 = scatter bf16 to (b,h,n,hd); 2 = +bias +resid -> f32;
//      3 = gelu(x+bias) -> bf16; 4 = scatter bf16 transposed to (b,h,hd,n).
// ---------------------------------------------------------------------------
template <int EPI>
__global__ __launch_bounds__(256) void gemm_wt_kernel(
    const bf16_t* __restrict__ A, const bf16_t* __restrict__ W,
    float* __restrict__ outf, bf16_t* __restrict__ outb,
    const float* __restrict__ bias, const float* __restrict__ resid,
    int M, int N, int K) {
  __shared__ bf16_t As[128 * 40];  // [m][k]
  __shared__ bf16_t Bs[64 * 40];   // [n][k]
  const int m0 = blockIdx.y * 128;
  const int n0 = blockIdx.x * 64;
  const int t = threadIdx.x;
  const int w = t >> 5, lane = t & 31;
  const int wm = (w >> 1) * 32, wn = (w & 1) * 32;
  v8f acc[2][2];
#pragma unroll
  for (int i = 0; i < 2; ++i)
#pragma unroll
    for (int j = 0; j < 2; ++j) acc[i][j] = v8f_zero();

  for (int k0 = 0; k0 < K; k0 += 32) {
    for (int i = t; i < 128 * 4; i += 256) {
      const int r = i >> 2, ch = i & 3;
      *(uint4*)&As[r * 40 + ch * 8] =
          *(const uint4*)&A[(size_t)(m0 + r) * K + k0 + ch * 8];
    }
    for (int i = t; i < 64 * 4; i += 256) {
      const int r = i >> 2, ch = i & 3;
      *(uint4*)&Bs[r * 40 + ch * 8] =
          *(const uint4*)&W[(size_t)(n0 + r) * K + k0 + ch * 8];
    }
    if (k0 + 32 < K) {  // CDNA5 speculative prefetch of the next K-slice
      __builtin_prefetch(&A[(size_t)(m0 + (t & 127)) * K + k0 + 32], 0, 1);
      __builtin_prefetch(&W[(size_t)(n0 + (t & 63)) * K + k0 + 32], 0, 1);
    }
    __syncthreads();
    v16bf a[2], b[2];
#pragma unroll
    for (int i = 0; i < 2; ++i) a[i] = frag_a_rm(&As[(wm + 16 * i) * 40], 40);
#pragma unroll
    for (int j = 0; j < 2; ++j) b[j] = frag_b_nk(&Bs[(wn + 16 * j) * 40], 40);
#pragma unroll
    for (int i = 0; i < 2; ++i)
#pragma unroll
      for (int j = 0; j < 2; ++j) acc[i][j] = wmma_bf16(a[i], b[j], acc[i][j]);
    __syncthreads();
  }

  const int colB = lane & 15, rowB = (lane >> 4) * 8;
#pragma unroll
  for (int i = 0; i < 2; ++i)
#pragma unroll
    for (int j = 0; j < 2; ++j)
#pragma unroll
      for (int r = 0; r < 8; ++r) {
        const int m = m0 + wm + 16 * i + rowB + r;
        const int n = n0 + wn + 16 * j + colB;
        float v = acc[i][j][r];
        if (EPI == 1) {  // (b*1024+ns, h*64+hd) -> (b,h,ns,hd)
          const int bb_ = m >> 10, ns = m & 1023;
          const int h = n >> 6, hd = n & 63;
          outb[(((size_t)bb_ * 8 + h) * 1024 + ns) * 64 + hd] = (bf16_t)v;
        } else if (EPI == 2) {
          v += bias[n] + resid[(size_t)m * N + n];
          outf[(size_t)m * N + n] = v;
        } else if (EPI == 3) {
          v += bias[n];
          v = 0.5f * v * (1.0f + erff(v * 0.70710678118654752f));  // exact GELU
          outb[(size_t)m * N + n] = (bf16_t)v;
        } else if (EPI == 4) {  // (b,h,hd,ns) transposed scatter (for V)
          const int bb_ = m >> 10, ns = m & 1023;
          const int h = n >> 6, hd = n & 63;
          outb[(((size_t)bb_ * 8 + h) * 64 + hd) * 1024 + ns] = (bf16_t)v;
        }
      }
}

// ---------------------------------------------------------------------------
// Feature projection: out(b,h,n,f) = elu(q(b,h,n,:) @ proj[h]) + 1
// A: (1024x64) row-major bf16 per head; B: projT[h] (128x64) = [n=f][k=hd].
// Optionally also stores the transposed map (b,h,f,n) (used for Kp).
// ---------------------------------------------------------------------------
__global__ __launch_bounds__(256) void feat_kernel(const bf16_t* __restrict__ Q,
                                                   const bf16_t* __restrict__ PT,
                                                   bf16_t* __restrict__ out,
                                                   bf16_t* __restrict__ outT) {
  __shared__ bf16_t As[128 * 40];  // [m][k]
  __shared__ bf16_t Bs[64 * 40];   // [n=f][k=hd]
  const int n0 = blockIdx.x * 64;   // F tile
  const int m0 = blockIdx.y * 128;  // seq tile
  const int bh = blockIdx.z;
  const int h = bh & 7;
  const bf16_t* Ah = Q + (size_t)bh * 1024 * 64;
  const bf16_t* Ph = PT + (size_t)h * 128 * 64;
  const int t = threadIdx.x;
  const int w = t >> 5, lane = t & 31;
  const int wm = (w >> 1) * 32, wn = (w & 1) * 32;
  v8f acc[2][2];
#pragma unroll
  for (int i = 0; i < 2; ++i)
#pragma unroll
    for (int j = 0; j < 2; ++j) acc[i][j] = v8f_zero();

  for (int k0 = 0; k0 < 64; k0 += 32) {
    for (int i = t; i < 128 * 4; i += 256) {
      const int r = i >> 2, ch = i & 3;
      *(uint4*)&As[r * 40 + ch * 8] =
          *(const uint4*)&Ah[(size_t)(m0 + r) * 64 + k0 + ch * 8];
    }
    {  // 64 rows (f) x 32 halfs (hd) = 256 16B chunks, one per thread
      const int r = t >> 2, ch = t & 3;
      *(uint4*)&Bs[r * 40 + ch * 8] =
          *(const uint4*)&Ph[(size_t)(n0 + r) * 64 + k0 + ch * 8];
    }
    __syncthreads();
    v16bf a[2], b[2];
#pragma unroll
    for (int i = 0; i < 2; ++i) a[i] = frag_a_rm(&As[(wm + 16 * i) * 40], 40);
#pragma unroll
    for (int j = 0; j < 2; ++j) b[j] = frag_b_nk(&Bs[(wn + 16 * j) * 40], 40);
#pragma unroll
    for (int i = 0; i < 2; ++i)
#pragma unroll
      for (int j = 0; j < 2; ++j) acc[i][j] = wmma_bf16(a[i], b[j], acc[i][j]);
    __syncthreads();
  }

  const int colB = lane & 15, rowB = (lane >> 4) * 8;
#pragma unroll
  for (int i = 0; i < 2; ++i)
#pragma unroll
    for (int j = 0; j < 2; ++j)
#pragma unroll
      for (int r = 0; r < 8; ++r) {
        const int m = m0 + wm + 16 * i + rowB + r;
        const int n = n0 + wn + 16 * j + colB;
        float v = acc[i][j][r];
        v = (v > 0.0f) ? v + 1.0f : expf(v);  // ELU + 1
        const bf16_t bv = (bf16_t)v;
        out[((size_t)bh * 1024 + m) * 128 + n] = bv;
        if (outT) outT[((size_t)bh * 128 + n) * 1024 + m] = bv;
      }
}

// ---------------------------------------------------------------------------
// Phase 1: per (b,h,chunk): Skv_local = Kp_c^T @ V_c (F=128 x HD=64, K=64) and
// per-feature column sums ksum_local[F]. Reads KpT (b,h,f,n) and VT (b,h,d,n),
// so both LDS tiles are K(=seq)-contiguous.
// ---------------------------------------------------------------------------
__global__ __launch_bounds__(256) void phase1_kernel(
    const bf16_t* __restrict__ KPT, const bf16_t* __restrict__ VT,
    float* __restrict__ skv, float* __restrict__ ksum) {
  __shared__ bf16_t kpt_s[128 * 72];  // [f][n]
  __shared__ bf16_t vt_s[64 * 72];    // [d][n]
  const int c = blockIdx.x, bh = blockIdx.y;
  const int t = threadIdx.x;
  const int w = t >> 5, lane = t & 31;
  const int wm = (w >> 1) * 32, wn = (w & 1) * 32;

  for (int i = t; i < 128 * 8; i += 256) {
    const int r = i >> 3, ch = i & 7;
    *(uint4*)&kpt_s[r * 72 + ch * 8] =
        *(const uint4*)&KPT[((size_t)bh * 128 + r) * 1024 + c * 64 + ch * 8];
  }
  for (int i = t; i < 64 * 8; i += 256) {
    const int r = i >> 3, ch = i & 7;
    *(uint4*)&vt_s[r * 72 + ch * 8] =
        *(const uint4*)&VT[((size_t)bh * 64 + r) * 1024 + c * 64 + ch * 8];
  }
  __syncthreads();

  if (t < 128) {  // per-feature chunk sums (contiguous row reads)
    float s = 0.0f;
    for (int nn = 0; nn < 64; ++nn) s += (float)kpt_s[t * 72 + nn];
    ksum[((size_t)bh * 16 + c) * 128 + t] = s;
  }

  v8f acc[2][2];
#pragma unroll
  for (int i = 0; i < 2; ++i)
#pragma unroll
    for (int j = 0; j < 2; ++j) acc[i][j] = v8f_zero();
#pragma unroll
  for (int ks = 0; ks < 2; ++ks) {
    const int k0 = ks * 32;
    v16bf a[2], b[2];
#pragma unroll
    for (int i = 0; i < 2; ++i) a[i] = frag_a_rm(&kpt_s[(wm + 16 * i) * 72 + k0], 72);
#pragma unroll
    for (int j = 0; j < 2; ++j) b[j] = frag_b_nk(&vt_s[(wn + 16 * j) * 72 + k0], 72);
#pragma unroll
    for (int i = 0; i < 2; ++i)
#pragma unroll
      for (int j = 0; j < 2; ++j) acc[i][j] = wmma_bf16(a[i], b[j], acc[i][j]);
  }

  const int colB = lane & 15, rowB = (lane >> 4) * 8;
  const size_t ob = ((size_t)bh * 16 + c) * 8192;
#pragma unroll
  for (int i = 0; i < 2; ++i)
#pragma unroll
    for (int j = 0; j < 2; ++j)
#pragma unroll
      for (int r = 0; r < 8; ++r) {
        const int f = wm + 16 * i + rowB + r;
        const int d = wn + 16 * j + colB;
        skv[ob + (size_t)f * 64 + d] = acc[i][j][r];
      }
}

// ---------------------------------------------------------------------------
// Phase 2: exclusive prefix over the 16 chunk states per (b,h).
// skv is [f][d]; the prefix is written TRANSPOSED as [d][f] bf16 so phase 3
// can read it as a K(=f)-contiguous B tile.
// ---------------------------------------------------------------------------
__global__ __launch_bounds__(256) void phase2_kernel(
    const float* __restrict__ skv, const float* __restrict__ ksuml,
    bf16_t* __restrict__ spref, float* __restrict__ ksump) {
  const int bh = blockIdx.x, t = threadIdx.x;
  float run[32];
#pragma unroll
  for (int j = 0; j < 32; ++j) run[j] = 0.0f;
  float runk = 0.0f;
  for (int c = 0; c < 16; ++c) {
    const size_t base = ((size_t)bh * 16 + c) * 8192;
#pragma unroll
    for (int j = 0; j < 32; ++j) {
      const int idx = t + j * 256;
      const int f = idx >> 6, d = idx & 63;
      spref[base + (size_t)d * 128 + f] = (bf16_t)run[j];
      run[j] += skv[base + idx];
    }
    if (t < 128) {
      const size_t kb = ((size_t)bh * 16 + c) * 128 + t;
      ksump[kb] = runk;
      runk += ksuml[kb];
    }
  }
}

// ---------------------------------------------------------------------------
// Phase 3: per (b,h,chunk), 4 waves:
//   att = Qp @ S_prefix + tril(Qp @ Kp^T) @ V        (all WMMA)
//   denom = Qp . ksum_prefix + rowsum(tril scores)   (shfl_xor reduction)
// ---------------------------------------------------------------------------
__global__ __launch_bounds__(128) void phase3_kernel(
    const bf16_t* __restrict__ QP, const bf16_t* __restrict__ KP,
    const bf16_t* __restrict__ VT, const bf16_t* __restrict__ SPT,
    const float* __restrict__ KSP, float* __restrict__ att,
    float* __restrict__ denom) {
  __shared__ bf16_t qp_s[64 * 136];  // [n][f]
  __shared__ bf16_t kv_u[64 * 136];  // Kp [mm][f], then S_prefixT [d][f]
  __shared__ bf16_t vt_s[64 * 72];   // [d][mm]
  __shared__ bf16_t sc_s[64 * 72];   // masked scores [n][mm]
  __shared__ float den_s[64];
  __shared__ float den_p[64 * 4];
  __shared__ float ksp_s[128];

  const int c = blockIdx.x, bh = blockIdx.y;
  const int t = threadIdx.x;
  const int w = t >> 5, lane = t & 31;
  const int wm = (w >> 1) * 32, wn = (w & 1) * 32;
  const size_t rowbase = (size_t)bh * 1024 + (size_t)c * 64;
  const size_t cbase = (size_t)bh * 16 + c;

  for (int i = t; i < 64 * 16; i += 128) {
    const int r = i >> 4, ch = i & 15;
    *(uint4*)&qp_s[r * 136 + ch * 8] =
        *(const uint4*)&QP[(rowbase + r) * 128 + ch * 8];
    *(uint4*)&kv_u[r * 136 + ch * 8] =
        *(const uint4*)&KP[(rowbase + r) * 128 + ch * 8];
  }
  for (int i = t; i < 64 * 8; i += 128) {
    const int r = i >> 3, ch = i & 7;
    *(uint4*)&vt_s[r * 72 + ch * 8] =
        *(const uint4*)&VT[((size_t)bh * 64 + r) * 1024 + c * 64 + ch * 8];
  }
  ksp_s[t] = KSP[cbase * 128 + t];
  __syncthreads();

  if (t < 64) {  // denom init: qp[nn] . ksum_prefix
    float s = 0.0f;
    for (int f = 0; f < 128; ++f) s += (float)qp_s[t * 136 + f] * ksp_s[f];
    den_s[t] = s;
  }

  // scores = Qp @ Kp^T (K = 128): A=[n][f], B=[mm][f], both K-contiguous.
  v8f sacc[2][2];
#pragma unroll
  for (int i = 0; i < 2; ++i)
#pragma unroll
    for (int j = 0; j < 2; ++j) sacc[i][j] = v8f_zero();
#pragma unroll
  for (int ks = 0; ks < 4; ++ks) {
    const int k0 = ks * 32;
    v16bf a[2], b[2];
#pragma unroll
    for (int i = 0; i < 2; ++i)
      a[i] = frag_a_rm(&qp_s[(wm + 16 * i) * 136 + k0], 136);
#pragma unroll
    for (int j = 0; j < 2; ++j)
      b[j] = frag_b_nk(&kv_u[(wn + 16 * j) * 136 + k0], 136);
#pragma unroll
    for (int i = 0; i < 2; ++i)
#pragma unroll
      for (int j = 0; j < 2; ++j) sacc[i][j] = wmma_bf16(a[i], b[j], sacc[i][j]);
  }

  const int colB = lane & 15, rowB = (lane >> 4) * 8;
#pragma unroll
  for (int i = 0; i < 2; ++i)
#pragma unroll
    for (int j = 0; j < 2; ++j)
#pragma unroll
      for (int r = 0; r < 8; ++r) {
        const int row = wm + 16 * i + rowB + r;  // query index
        const int col = wn + 16 * j + colB;      // key index
        float v = sacc[i][j][r];
        if (col > row) v = 0.0f;  // causal mask (inclusive)
        sc_s[row * 72 + col] = (bf16_t)v;
        float s = v;  // deterministic rowsum across the 16-lane column group
        s += __shfl_xor(s, 1);
        s += __shfl_xor(s, 2);
        s += __shfl_xor(s, 4);
        s += __shfl_xor(s, 8);
        if (colB == 0) den_p[row * 4 + (wn >> 4) + j] = s;
      }
  __syncthreads();

  if (t < 64)
    den_s[t] += den_p[t * 4 + 0] + den_p[t * 4 + 1] + den_p[t * 4 + 2] +
                den_p[t * 4 + 3];
  // stage S_prefixT (64 d x 128 f) into kv_u (Kp reads are done)
  for (int i = t; i < 64 * 16; i += 128) {
    const int r = i >> 4, ch = i & 15;
    *(uint4*)&kv_u[r * 136 + ch * 8] =
        *(const uint4*)&SPT[cbase * 8192 + (size_t)r * 128 + ch * 8];
  }
  __syncthreads();

  v8f acc[2][2];
#pragma unroll
  for (int i = 0; i < 2; ++i)
#pragma unroll
    for (int j = 0; j < 2; ++j) acc[i][j] = v8f_zero();
  // GEMM1: Qp (64x128) @ S_prefix (128x64); B tile [d][f] K-contiguous.
#pragma unroll
  for (int ks = 0; ks < 4; ++ks) {
    const int k0 = ks * 32;
    v16bf a[2], b[2];
#pragma unroll
    for (int i = 0; i < 2; ++i)
      a[i] = frag_a_rm(&qp_s[(wm + 16 * i) * 136 + k0], 136);
#pragma unroll
    for (int j = 0; j < 2; ++j)
      b[j] = frag_b_nk(&kv_u[(wn + 16 * j) * 136 + k0], 136);
#pragma unroll
    for (int i = 0; i < 2; ++i)
#pragma unroll
      for (int j = 0; j < 2; ++j) acc[i][j] = wmma_bf16(a[i], b[j], acc[i][j]);
  }
  // GEMM3: tril-scores (64x64) @ V (64x64); B tile [d][mm] K-contiguous.
#pragma unroll
  for (int ks = 0; ks < 2; ++ks) {
    const int k0 = ks * 32;
    v16bf a[2], b[2];
#pragma unroll
    for (int i = 0; i < 2; ++i) a[i] = frag_a_rm(&sc_s[(wm + 16 * i) * 72 + k0], 72);
#pragma unroll
    for (int j = 0; j < 2; ++j) b[j] = frag_b_nk(&vt_s[(wn + 16 * j) * 72 + k0], 72);
#pragma unroll
    for (int i = 0; i < 2; ++i)
#pragma unroll
      for (int j = 0; j < 2; ++j) acc[i][j] = wmma_bf16(a[i], b[j], acc[i][j]);
  }

#pragma unroll
  for (int i = 0; i < 2; ++i)
#pragma unroll
    for (int j = 0; j < 2; ++j)
#pragma unroll
      for (int r = 0; r < 8; ++r) {
        const int row = wm + 16 * i + rowB + r;
        const int col = wn + 16 * j + colB;
        att[(rowbase + row) * 64 + col] = acc[i][j][r];
      }
  if (t < 64) denom[rowbase + t] = den_s[t];
}

// ---------------------------------------------------------------------------
// Channel-mixing conv over n (64 in/out channels), f32. Each thread covers 32
// contiguous output positions; the sliding input window is register-resident.
// ---------------------------------------------------------------------------
template <int KS>
__global__ __launch_bounds__(256) void conv_kernel(const float* __restrict__ in,
                                                   float* __restrict__ out,
                                                   const float* __restrict__ w,
                                                   const float* __restrict__ bias) {
  constexpr int PAD = KS / 2;
  constexpr int W_ = 128 + 2 * PAD;
  __shared__ float xt[W_ * 64];  // [pos][ch]
  const int n0 = blockIdx.x * 128;
  const int bh = blockIdx.y;
  const int t = threadIdx.x;
  const size_t base = (size_t)bh * 1024;
  for (int i = t; i < W_ * 64; i += 256) {
    const int pos = i >> 6, ch = i & 63;
    const int n = n0 + pos - PAD;
    xt[i] = (n >= 0 && n < 1024) ? in[(base + n) * 64 + ch] : 0.0f;
  }
  __syncthreads();
  const int co = t & 63;
  const int p0 = (t >> 6) * 32;  // 0,32,64,96
  float acc[32];
  const float b0 = bias[co];
#pragma unroll
  for (int p = 0; p < 32; ++p) acc[p] = b0;
  const float* wr = w + (size_t)co * 64 * KS;
  for (int ci = 0; ci < 64; ++ci) {
    float xin[32 + KS - 1];
#pragma unroll
    for (int j = 0; j < 32 + KS - 1; ++j) xin[j] = xt[(p0 + j) * 64 + ci];
#pragma unroll
    for (int tt = 0; tt < KS; ++tt) {
      const float wv = wr[ci * KS + tt];
#pragma unroll
      for (int p = 0; p < 32; ++p) acc[p] = fmaf(wv, xin[p + tt], acc[p]);
    }
  }
#pragma unroll
  for (int p = 0; p < 32; ++p)
    out[(base + n0 + p0 + p) * 64 + co] = acc[p];
}

// ---------------------------------------------------------------------------
// Divide by denom+eps and merge heads: (b,h,n,d) -> (b*n, h*64+d) bf16
// ---------------------------------------------------------------------------
__global__ __launch_bounds__(256) void divmerge_kernel(
    const float* __restrict__ att, const float* __restrict__ den,
    bf16_t* __restrict__ out) {
  const size_t i = (size_t)blockIdx.x * 256 + threadIdx.x;  // 2^20 total
  const int d = (int)(i & 63);
  const int n = (int)((i >> 6) & 1023);
  const int h = (int)((i >> 16) & 7);
  const int b = (int)(i >> 19);
  const float v = att[i] / (den[i >> 6] + 1e-8f);
  out[((size_t)b * 1024 + n) * 512 + h * 64 + d] = (bf16_t)v;
}

// ---------------------------------------------------------------------------
// LayerNorm over 512 cols; one block per row (2 elements / thread).
// ---------------------------------------------------------------------------
__global__ __launch_bounds__(256) void ln_kernel(const float* __restrict__ in,
                                                 const float* __restrict__ g,
                                                 const float* __restrict__ bb,
                                                 float* __restrict__ outf,
                                                 bf16_t* __restrict__ outb) {
  __shared__ float red[256];
  const int row = blockIdx.x, t = threadIdx.x;
  const float* x = in + (size_t)row * 512;
  const float v0 = x[t], v1 = x[t + 256];
  red[t] = v0 + v1;
  __syncthreads();
  for (int o = 128; o > 0; o >>= 1) {
    if (t < o) red[t] += red[t + o];
    __syncthreads();
  }
  const float mu = red[0] * (1.0f / 512.0f);
  __syncthreads();
  red[t] = v0 * v0 + v1 * v1;
  __syncthreads();
  for (int o = 128; o > 0; o >>= 1) {
    if (t < o) red[t] += red[t + o];
    __syncthreads();
  }
  const float var = red[0] * (1.0f / 512.0f) - mu * mu;
  const float rs = rsqrtf(var + 1e-5f);
  const float y0 = (v0 - mu) * rs * g[t] + bb[t];
  const float y1 = (v1 - mu) * rs * g[t + 256] + bb[t + 256];
  outf[(size_t)row * 512 + t] = y0;
  outf[(size_t)row * 512 + t + 256] = y1;
  if (outb) {
    outb[(size_t)row * 512 + t] = (bf16_t)y0;
    outb[(size_t)row * 512 + t + 256] = (bf16_t)y1;
  }
}

// ---------------------------------------------------------------------------
extern "C" void kernel_launch(void* const* d_in, const int* in_sizes, int n_in,
                              void* d_out, int out_size, void* d_ws, size_t ws_size,
                              hipStream_t stream) {
  (void)in_sizes; (void)n_in; (void)out_size; (void)ws_size;
  const float* x    = (const float*)d_in[0];
  const float* y    = (const float*)d_in[1];
  const float* Wq   = (const float*)d_in[2];
  const float* Wk   = (const float*)d_in[3];
  const float* Wv   = (const float*)d_in[4];
  const float* proj = (const float*)d_in[5];
  const float* cw3  = (const float*)d_in[6];
  const float* cb3  = (const float*)d_in[7];
  const float* cw5  = (const float*)d_in[8];
  const float* cb5  = (const float*)d_in[9];
  const float* cw7  = (const float*)d_in[10];
  const float* cb7  = (const float*)d_in[11];
  const float* cw9  = (const float*)d_in[12];
  const float* cb9  = (const float*)d_in[13];
  const float* ng   = (const float*)d_in[14];
  const float* nbv  = (const float*)d_in[15];
  const float* w1   = (const float*)d_in[16];
  const float* b1   = (const float*)d_in[17];
  const float* w2   = (const float*)d_in[18];
  const float* b2   = (const float*)d_in[19];
  const float* ow   = (const float*)d_in[20];
  const float* ob   = (const float*)d_in[21];
  float* outp = (float*)d_out;

  char* p = (char*)d_ws;
  auto alloc = [&](size_t bytes) {
    char* r = p;
    p += (bytes + 255) & ~(size_t)255;
    return r;
  };
  bf16_t* xbf     = (bf16_t*)alloc(2048ull * 512 * 2);
  bf16_t* ybf     = (bf16_t*)alloc(2048ull * 512 * 2);
  bf16_t* wq_b    = (bf16_t*)alloc(512ull * 512 * 2);
  bf16_t* wk_b    = (bf16_t*)alloc(512ull * 512 * 2);
  bf16_t* wv_b    = (bf16_t*)alloc(512ull * 512 * 2);
  bf16_t* projT_b = (bf16_t*)alloc(8ull * 128 * 64 * 2);
  bf16_t* ow_b    = (bf16_t*)alloc(512ull * 512 * 2);
  bf16_t* w1_b    = (bf16_t*)alloc(2048ull * 512 * 2);
  bf16_t* w2_b    = (bf16_t*)alloc(512ull * 2048 * 2);
  bf16_t* q_b     = (bf16_t*)alloc(16ull * 1024 * 64 * 2);
  bf16_t* k_b     = (bf16_t*)alloc(16ull * 1024 * 64 * 2);
  bf16_t* vT_b    = (bf16_t*)alloc(16ull * 64 * 1024 * 2);
  bf16_t* qp_b    = (bf16_t*)alloc(16ull * 1024 * 128 * 2);
  bf16_t* kp_b    = (bf16_t*)alloc(16ull * 1024 * 128 * 2);
  bf16_t* kpT_b   = (bf16_t*)alloc(16ull * 128 * 1024 * 2);
  float*  skv     = (float*)alloc(16ull * 16 * 8192 * 4);
  float*  ksuml   = (float*)alloc(16ull * 16 * 128 * 4);
  bf16_t* spT_b   = (bf16_t*)alloc(16ull * 16 * 8192 * 2);
  float*  ksump   = (float*)alloc(16ull * 16 * 128 * 4);
  float*  att_a   = (float*)alloc(16ull * 1024 * 64 * 4);
  float*  att_b   = (float*)alloc(16ull * 1024 * 64 * 4);
  float*  denomb  = (float*)alloc(16ull * 1024 * 4);
  bf16_t* attm_b  = (bf16_t*)alloc(2048ull * 512 * 2);
  float*  h1      = (float*)alloc(2048ull * 512 * 4);
  float*  h1n     = (float*)alloc(2048ull * 512 * 4);
  bf16_t* h1n_b   = (bf16_t*)alloc(2048ull * 512 * 2);
  bf16_t* hid_b   = (bf16_t*)alloc(2048ull * 2048 * 2);
  float*  pre2    = (float*)alloc(2048ull * 512 * 4);

  const dim3 blk(256);
  auto C = [&](const float* s, bf16_t* d, int n) {
    cvt_kernel<<<(n + 255) / 256, blk, 0, stream>>>(s, d, n);
  };
  C(x, xbf, 2048 * 512);      C(y, ybf, 2048 * 512);
  C(Wq, wq_b, 512 * 512);     C(Wk, wk_b, 512 * 512);
  C(Wv, wv_b, 512 * 512);     C(ow, ow_b, 512 * 512);
  C(w1, w1_b, 2048 * 512);    C(w2, w2_b, 512 * 2048);
  cvt_tr_proj_kernel<<<(8 * 64 * 128) / 256, blk, 0, stream>>>(proj, projT_b);

  // QKV projections; V is stored transposed (b,h,hd,n) for K-contiguous tiles.
  gemm_wt_kernel<1><<<dim3(8, 16), blk, 0, stream>>>(xbf, wq_b, nullptr, q_b,
                                                     nullptr, nullptr, 2048, 512, 512);
  gemm_wt_kernel<1><<<dim3(8, 16), blk, 0, stream>>>(ybf, wk_b, nullptr, k_b,
                                                     nullptr, nullptr, 2048, 512, 512);
  gemm_wt_kernel<4><<<dim3(8, 16), blk, 0, stream>>>(ybf, wv_b, nullptr, vT_b,
                                                     nullptr, nullptr, 2048, 512, 512);
  // Feature maps: elu(.@proj)+1; Kp also stored transposed (b,h,f,n).
  feat_kernel<<<dim3(2, 8, 16), blk, 0, stream>>>(q_b, projT_b, qp_b, nullptr);
  feat_kernel<<<dim3(2, 8, 16), blk, 0, stream>>>(k_b, projT_b, kp_b, kpT_b);
  // Chunked causal linear attention
  phase1_kernel<<<dim3(16, 16), blk, 0, stream>>>(kpT_b, vT_b, skv, ksuml);
  phase2_kernel<<<dim3(16), blk, 0, stream>>>(skv, ksuml, spT_b, ksump);
  phase3_kernel<<<dim3(16, 16), dim3(128), 0, stream>>>(qp_b, kp_b, vT_b, spT_b,
                                                        ksump, att_a, denomb);
  // Conv stack (k = 3,5,7,9), ping-pong buffers
  conv_kernel<3><<<dim3(8, 16), blk, 0, stream>>>(att_a, att_b, cw3, cb3);
  conv_kernel<5><<<dim3(8, 16), blk, 0, stream>>>(att_b, att_a, cw5, cb5);
  conv_kernel<7><<<dim3(8, 16), blk, 0, stream>>>(att_a, att_b, cw7, cb7);
  conv_kernel<9><<<dim3(8, 16), blk, 0, stream>>>(att_b, att_a, cw9, cb9);
  // Divide by denom, merge heads
  divmerge_kernel<<<dim3(4096), blk, 0, stream>>>(att_a, denomb, attm_b);
  // Out projection + residual, LN, MLP, LN
  gemm_wt_kernel<2><<<dim3(8, 16), blk, 0, stream>>>(attm_b, ow_b, h1, nullptr,
                                                     ob, x, 2048, 512, 512);
  ln_kernel<<<dim3(2048), blk, 0, stream>>>(h1, ng, nbv, h1n, h1n_b);
  gemm_wt_kernel<3><<<dim3(32, 16), blk, 0, stream>>>(h1n_b, w1_b, nullptr, hid_b,
                                                      b1, nullptr, 2048, 2048, 512);
  gemm_wt_kernel<2><<<dim3(8, 16), blk, 0, stream>>>(hid_b, w2_b, pre2, nullptr,
                                                     b2, h1n, 2048, 512, 2048);
  ln_kernel<<<dim3(2048), blk, 0, stream>>>(pre2, ng, nbv, outp, nullptr);
}
